// PointNetEncoder_mac_63007170232723
// MI455X (gfx1250) — compile-verified, matrix operand order checked by reading, not run on hardware
//
#include <hip/hip_runtime.h>
#include <cmath>

// ---------------------------------------------------------------------------
// VN-PointNet encoder for MI455X (gfx1250, wave32, WMMA).
//
// Activation layout: row-major [C, M] with column m = (b*3 + v)*S + s.
// All vn_linear ops are GEMMs Y[Cout,M] = W[Cout,Cin] * X[Cin,M] on a tiled
// v_wmma_f32_16x16x32_f16 kernel (f32 in/out, f16 MACs, f32 accumulation).
// Every M is a multiple of 128 (fc stage padded 48 -> 128 cols) so the GEMM
// column path has zero bounds checks. The K loop is split into a clamp-free
// main loop (strength-reduced pointers, packed cvt_pk_rtz f16 stores) and a
// single guarded tail tile.
// ---------------------------------------------------------------------------

typedef __attribute__((ext_vector_type(16))) _Float16 v16h;
typedef __attribute__((ext_vector_type(8)))  _Float16 v8h;
typedef __attribute__((ext_vector_type(2)))  _Float16 v2h;
typedef __attribute__((ext_vector_type(2)))  __fp16   v2hh;   // cvt_pkrtz ret type
typedef __attribute__((ext_vector_type(8)))  float    v8f;

static __device__ __forceinline__ v2h pk2(float lo, float hi) {
  v2hh t = __builtin_amdgcn_cvt_pkrtz(lo, hi);   // v_cvt_pk_rtz_f16_f32
  return __builtin_bit_cast(v2h, t);
}

#define EPS_V 1e-6f
#define BN_EPS 1e-5f
#define KNN 20
#define NB 16
#define NPTS 2048

// ---------------------------------------------------------------------------
// 1) Brute-force KNN (top-20 by similarity), one thread per point.
// ---------------------------------------------------------------------------
__global__ void knn_kernel(const float* __restrict__ x, int* __restrict__ idx) {
  int t = blockIdx.x * blockDim.x + threadIdx.x;
  if (t >= NB * NPTS) return;
  int b = t / NPTS, n = t % NPTS;
  const float* xb = x + (size_t)b * 3 * NPTS;
  float px = xb[n], py = xb[NPTS + n], pz = xb[2 * NPTS + n];
  float psq = px * px + py * py + pz * pz;
  float bestv[KNN];
  int   besti[KNN];
  for (int i = 0; i < KNN; ++i) { bestv[i] = -3.4e38f; besti[i] = 0; }
  for (int j = 0; j < NPTS; ++j) {
    float qx = xb[j], qy = xb[NPTS + j], qz = xb[2 * NPTS + j];
    float sim = 2.f * (px * qx + py * qy + pz * qz) - psq
                - (qx * qx + qy * qy + qz * qz);
    if (sim > bestv[KNN - 1]) {
      int p = KNN - 1;
      while (p > 0 && bestv[p - 1] < sim) {
        bestv[p] = bestv[p - 1]; besti[p] = besti[p - 1]; --p;
      }
      bestv[p] = sim; besti[p] = j;
    }
  }
  int* o = idx + (size_t)t * KNN;
  for (int i = 0; i < KNN; ++i) o[i] = besti[i];
}

// ---------------------------------------------------------------------------
// 2) Build graph features (nbr-x, x, cross(nbr,x)) into [3, M1] layout.
// ---------------------------------------------------------------------------
__global__ void feat_kernel(const float* __restrict__ x, const int* __restrict__ idx,
                            float* __restrict__ feat, int M1) {
  int t = blockIdx.x * blockDim.x + threadIdx.x;
  int total = NB * NPTS * KNN;
  if (t >= total) return;
  int kk = t % KNN;
  int n  = (t / KNN) % NPTS;
  int b  = t / (KNN * NPTS);
  const float* xb = x + (size_t)b * 3 * NPTS;
  float p0 = xb[n], p1 = xb[NPTS + n], p2 = xb[2 * NPTS + n];
  int j = idx[(size_t)t];
  float q0 = xb[j], q1 = xb[NPTS + j], q2 = xb[2 * NPTS + j];
  float d0 = q0 - p0, d1 = q1 - p1, d2 = q2 - p2;       // nbr - x
  float c0 = q1 * p2 - q2 * p1;                         // cross(nbr, x)
  float c1 = q2 * p0 - q0 * p2;
  float c2 = q0 * p1 - q1 * p0;
  const int S1 = NPTS * KNN;
  size_t base = (size_t)(b * 3) * S1 + (size_t)n * KNN + kk;
  float dv[3] = {d0, d1, d2}, pv[3] = {p0, p1, p2}, cv[3] = {c0, c1, c2};
  for (int v = 0; v < 3; ++v) {
    size_t m = base + (size_t)v * S1;
    feat[m]                  = dv[v];
    feat[(size_t)M1 + m]     = pv[v];
    feat[2 * (size_t)M1 + m] = cv[v];
  }
}

// ---------------------------------------------------------------------------
// 3) Tiled WMMA GEMM: Y[Cout,M] = W[Cout,Cin] * X[Cin,M].
//    Block = 128 threads = 4 waves; block tile = 16 rows x 128 cols; each
//    wave owns two 16x16 tiles (A fragment reused). M % 128 == 0.
//    X tile stored TRANSPOSED in LDS (fragments = 2x ds_load_b128 each).
//    Main K loop has no bounds checks; one guarded tail tile handles Cin%32.
// ---------------------------------------------------------------------------
__global__ void gemm_wmma_f16(const float* __restrict__ W, const float* __restrict__ X,
                              float* __restrict__ Y, int Cout, int Cin, int M) {
  // XsT[col][k]: 128 cols x 32 k, row stride 40 halves (80 B = 16 B aligned,
  // 16 consecutive rows hit 16 distinct bank groups).
  __shared__ _Float16 XsT[128][40];
  __shared__ _Float16 Ws[16][32];
  const int tid  = threadIdx.x;
  const int lane = tid & 31;
  const int wave = tid >> 5;                  // 0..3
  const int row0 = blockIdx.y * 16;
  const int col0 = blockIdx.x * 128;
  const int lr   = lane & 15;
  const int half = lane >> 4;                 // 0 or 1

  // staging geometry: each thread stages a PAIR of adjacent K rows x 4 cols
  // per pass; pair converts fuse into v_cvt_pk_rtz + one ds_store_b32.
  const int krow = wave;                      // 0..3 -> row pair base 2*krow
  const int cc   = (tid & 31) * 4;            // column group
  const float* xp = X + (size_t)(2 * krow) * M + col0 + cc;

  // W staging: thread covers rows wr+{0,4,8,12} at fixed k index wk.
  const int wr = wave, wk = tid & 31;
  const float* wptr[4];
  bool wok[4];
#pragma unroll
  for (int p = 0; p < 4; ++p) {
    int gr = row0 + wr + p * 4;
    wok[p] = gr < Cout;
    int grc = wok[p] ? gr : (Cout - 1);
    wptr[p] = W + (size_t)grc * Cin + wk;
  }

  v8f acc0 = {}, acc1 = {};
  const int kfull = Cin & ~31;

  for (int k0 = 0; k0 < kfull; k0 += 32) {
    // ---- stage X tile: 4 passes x (row pair x 128 cols), no checks.
#pragma unroll
    for (int p = 0; p < 4; ++p) {
      const float* r0 = xp + (size_t)(p * 8) * M;
      const float4 a = *(const float4*)r0;
      const float4 b = *(const float4*)(r0 + M);
      int dw = p * 8 + 2 * krow;              // even k of the pair
      *(v2h*)&XsT[cc + 0][dw] = pk2(a.x, b.x);
      *(v2h*)&XsT[cc + 1][dw] = pk2(a.y, b.y);
      *(v2h*)&XsT[cc + 2][dw] = pk2(a.z, b.z);
      *(v2h*)&XsT[cc + 3][dw] = pk2(a.w, b.w);
    }
    // ---- stage W tile: row-clamped only (k in range for full tiles).
#pragma unroll
    for (int p = 0; p < 4; ++p) {
      float v = wptr[p][k0];
      Ws[wr + p * 4][wk] = (_Float16)(wok[p] ? v : 0.f);
    }
    // prefetch next K tile of X (global_prefetch_b8)
    if (k0 + 32 < Cin)
      __builtin_prefetch(xp + (size_t)32 * M, 0, 1);
    __syncthreads();
    // ---- fragments: two contiguous 8-half runs each (ds_load_b128 pairs).
    v8h alo = *(const v8h*)&Ws[lr][half * 8];
    v8h ahi = *(const v8h*)&Ws[lr][16 + half * 8];
    v16h a = __builtin_shufflevector(alo, ahi, 0, 1, 2, 3, 4, 5, 6, 7,
                                     8, 9, 10, 11, 12, 13, 14, 15);
    const _Float16* xr0 = &XsT[wave * 32 + lr][0];
    const _Float16* xr1 = &XsT[wave * 32 + 16 + lr][0];
    v8h b0lo = *(const v8h*)&xr0[half * 8];
    v8h b0hi = *(const v8h*)&xr0[16 + half * 8];
    v8h b1lo = *(const v8h*)&xr1[half * 8];
    v8h b1hi = *(const v8h*)&xr1[16 + half * 8];
    v16h b0 = __builtin_shufflevector(b0lo, b0hi, 0, 1, 2, 3, 4, 5, 6, 7,
                                      8, 9, 10, 11, 12, 13, 14, 15);
    v16h b1 = __builtin_shufflevector(b1lo, b1hi, 0, 1, 2, 3, 4, 5, 6, 7,
                                      8, 9, 10, 11, 12, 13, 14, 15);
    acc0 = __builtin_amdgcn_wmma_f32_16x16x32_f16(false, a, false, b0, (short)0,
                                                  acc0, false, false);
    acc1 = __builtin_amdgcn_wmma_f32_16x16x32_f16(false, a, false, b1, (short)0,
                                                  acc1, false, false);
    __syncthreads();
    xp += (size_t)32 * M;
  }

  // ---- tail tile (Cin % 32 != 0 for every layer here): clamped loads.
  if (kfull < Cin) {
    const int k0 = kfull;
#pragma unroll
    for (int p = 0; p < 4; ++p) {
      int ke = p * 8 + 2 * krow;
      int gk0 = k0 + ke, gk1 = gk0 + 1;
      int gk0c = gk0 < Cin ? gk0 : Cin - 1;
      int gk1c = gk1 < Cin ? gk1 : Cin - 1;
      const float4 a = *(const float4*)(X + (size_t)gk0c * M + col0 + cc);
      const float4 b = *(const float4*)(X + (size_t)gk1c * M + col0 + cc);
      bool ok0 = gk0 < Cin, ok1 = gk1 < Cin;
      float ax = ok0 ? a.x : 0.f, ay = ok0 ? a.y : 0.f,
            az = ok0 ? a.z : 0.f, aw = ok0 ? a.w : 0.f;
      float bx = ok1 ? b.x : 0.f, by = ok1 ? b.y : 0.f,
            bz = ok1 ? b.z : 0.f, bw = ok1 ? b.w : 0.f;
      *(v2h*)&XsT[cc + 0][ke] = pk2(ax, bx);
      *(v2h*)&XsT[cc + 1][ke] = pk2(ay, by);
      *(v2h*)&XsT[cc + 2][ke] = pk2(az, bz);
      *(v2h*)&XsT[cc + 3][ke] = pk2(aw, bw);
    }
#pragma unroll
    for (int p = 0; p < 4; ++p) {
      int gk = k0 + wk;
      bool kok = gk < Cin;
      float v = wptr[p][kok ? k0 : -wk];      // clamp to column 0 if k OOB
      Ws[wr + p * 4][wk] = (_Float16)((wok[p] && kok) ? v : 0.f);
    }
    __syncthreads();
    v8h alo = *(const v8h*)&Ws[lr][half * 8];
    v8h ahi = *(const v8h*)&Ws[lr][16 + half * 8];
    v16h a = __builtin_shufflevector(alo, ahi, 0, 1, 2, 3, 4, 5, 6, 7,
                                     8, 9, 10, 11, 12, 13, 14, 15);
    const _Float16* xr0 = &XsT[wave * 32 + lr][0];
    const _Float16* xr1 = &XsT[wave * 32 + 16 + lr][0];
    v8h b0lo = *(const v8h*)&xr0[half * 8];
    v8h b0hi = *(const v8h*)&xr0[16 + half * 8];
    v8h b1lo = *(const v8h*)&xr1[half * 8];
    v8h b1hi = *(const v8h*)&xr1[16 + half * 8];
    v16h b0 = __builtin_shufflevector(b0lo, b0hi, 0, 1, 2, 3, 4, 5, 6, 7,
                                      8, 9, 10, 11, 12, 13, 14, 15);
    v16h b1 = __builtin_shufflevector(b1lo, b1hi, 0, 1, 2, 3, 4, 5, 6, 7,
                                      8, 9, 10, 11, 12, 13, 14, 15);
    acc0 = __builtin_amdgcn_wmma_f32_16x16x32_f16(false, a, false, b0, (short)0,
                                                  acc0, false, false);
    acc1 = __builtin_amdgcn_wmma_f32_16x16x32_f16(false, a, false, b1, (short)0,
                                                  acc1, false, false);
  }

  // D layout: VGPR r -> row r (lanes 0-15) / r+8 (lanes 16-31), col = lane&15.
#pragma unroll
  for (int r = 0; r < 8; ++r) {
    int gr = row0 + r + half * 8;
    if (gr < Cout) {
      size_t rowoff = (size_t)gr * M;
      Y[rowoff + col0 + wave * 32 + lr]      = acc0[r];
      Y[rowoff + col0 + wave * 32 + 16 + lr] = acc1[r];
    }
  }
}

// ---------------------------------------------------------------------------
// 4) VN BatchNorm statistics: per-channel mean/var of |q| over (b, s).
// ---------------------------------------------------------------------------
__global__ void bn_stats_kernel(const float* __restrict__ Q, float* __restrict__ stats,
                                int S, int M) {
  int c = blockIdx.x;
  const float* base = Q + (size_t)c * M;
  long total = (long)NB * S;
  float sum = 0.f, sq = 0.f;
  for (long t = threadIdx.x; t < total; t += blockDim.x) {
    long b = t / S, s = t % S;
    const float* p = base + (size_t)b * 3 * S + s;
    float q0 = p[0], q1 = p[(size_t)S], q2 = p[2 * (size_t)S];
    float nr = sqrtf(q0 * q0 + q1 * q1 + q2 * q2) + EPS_V;
    sum += nr; sq += nr * nr;
  }
  __shared__ float s1[256], s2[256];
  s1[threadIdx.x] = sum; s2[threadIdx.x] = sq;
  __syncthreads();
  for (int o = 128; o > 0; o >>= 1) {
    if ((int)threadIdx.x < o) { s1[threadIdx.x] += s1[threadIdx.x + o];
                                s2[threadIdx.x] += s2[threadIdx.x + o]; }
    __syncthreads();
  }
  if (threadIdx.x == 0) {
    float mean = s1[0] / (float)total;
    float var  = s2[0] / (float)total - mean * mean;
    stats[2 * c] = mean; stats[2 * c + 1] = var;
  }
}

// ---------------------------------------------------------------------------
// 5) Fused VN-BN (+ optional direction-gated leaky ReLU, slope 0). In-place OK.
// ---------------------------------------------------------------------------
__global__ void bn_lrelu_kernel(const float* __restrict__ Q, const float* __restrict__ D,
                                const float* __restrict__ stats,
                                const float* __restrict__ gamma, const float* __restrict__ beta,
                                float* __restrict__ Y, int C, int S, int M, int gated) {
  long t = (long)blockIdx.x * blockDim.x + threadIdx.x;
  long per_c = (long)NB * S;
  long total = (long)C * per_c;
  if (t >= total) return;
  int c = (int)(t / per_c);
  long rem = t % per_c;
  long b = rem / S, s = rem % S;
  size_t off = (size_t)c * M + (size_t)b * 3 * S + s;
  float q0 = Q[off], q1 = Q[off + (size_t)S], q2 = Q[off + 2 * (size_t)S];
  float nr = sqrtf(q0 * q0 + q1 * q1 + q2 * q2) + EPS_V;
  float mean = stats[2 * c], var = stats[2 * c + 1];
  float scale = ((nr - mean) * rsqrtf(var + BN_EPS) * gamma[c] + beta[c]) / nr;
  q0 *= scale; q1 *= scale; q2 *= scale;
  if (gated) {
    float d0 = D[off], d1 = D[off + (size_t)S], d2 = D[off + 2 * (size_t)S];
    float dot = q0 * d0 + q1 * d1 + q2 * d2;
    if (dot < 0.f) {
      float coef = dot / (d0 * d0 + d1 * d1 + d2 * d2 + EPS_V);
      q0 -= coef * d0; q1 -= coef * d1; q2 -= coef * d2;
    }
  }
  Y[off] = q0; Y[off + (size_t)S] = q1; Y[off + 2 * (size_t)S] = q2;
}

// ---------------------------------------------------------------------------
// 6) VN max-pool over the last L spatial elements (argmax of <x, d>).
// ---------------------------------------------------------------------------
__global__ void vn_maxpool_kernel(const float* __restrict__ X, const float* __restrict__ D,
                                  float* __restrict__ Y, int C, int Nout, int L,
                                  int M, int Mout) {
  int t = blockIdx.x * blockDim.x + threadIdx.x;
  int total = C * NB * Nout;
  if (t >= total) return;
  int c = t / (NB * Nout);
  int rem = t % (NB * Nout);
  int b = rem / Nout, n = rem % Nout;
  size_t S = (size_t)Nout * L;
  size_t in = (size_t)c * M + (size_t)b * 3 * S + (size_t)n * L;
  const float* xb = X + in;
  const float* db = D + in;
  float best = -3.4e38f; int bl = 0;
  for (int l = 0; l < L; ++l) {
    float dot = xb[l] * db[l] + xb[S + l] * db[S + l] + xb[2 * S + l] * db[2 * S + l];
    if (dot > best) { best = dot; bl = l; }
  }
  float* y = Y + (size_t)c * Mout + (size_t)b * 3 * Nout + n;
  y[0]                = xb[bl];
  y[(size_t)Nout]     = xb[S + bl];
  y[2 * (size_t)Nout] = xb[2 * S + bl];
}

// ---------------------------------------------------------------------------
// 7) Broadcast pooled STN vector s[21, ld] into rows [row0, row0+C) of Y.
// ---------------------------------------------------------------------------
__global__ void broadcast_kernel(const float* __restrict__ S, float* __restrict__ Y,
                                 int C, int row0, int M, int Sld) {
  long t = (long)blockIdx.x * blockDim.x + threadIdx.x;
  long total = (long)C * 48 * NPTS;
  if (t >= total) return;
  int c = (int)(t / (48L * NPTS));
  long rem = t % (48L * NPTS);
  int r = (int)(rem / NPTS), n = (int)(rem % NPTS);
  Y[(size_t)(row0 + c) * M + (size_t)r * NPTS + n] = S[c * Sld + r];
}

// ---------------------------------------------------------------------------
// 8) Append per-(c,b,v) mean over n into rows [C, 2C) of H (concat(h, mean)).
// ---------------------------------------------------------------------------
__global__ void mean_rows_kernel(float* __restrict__ H, int C, int M) {
  int t = blockIdx.x * blockDim.x + threadIdx.x;
  if (t >= C * 48) return;
  int c = t / 48, r = t % 48;
  const float* src = H + (size_t)c * M + (size_t)r * NPTS;
  float s = 0.f;
  for (int n = 0; n < NPTS; ++n) s += src[n];
  s /= (float)NPTS;
  float* dst = H + (size_t)(C + c) * M + (size_t)r * NPTS;
  for (int n = 0; n < NPTS; ++n) dst[n] = s;
}

// ---------------------------------------------------------------------------
// 9) Final: out[b, c*3+kv] = max_n sum_j H[c,(b,j,n)] * Z[kv,(b,j,n)].
// ---------------------------------------------------------------------------
__global__ void final_max_kernel(const float* __restrict__ H, const float* __restrict__ Z,
                                 float* __restrict__ out, int M) {
  int t = blockIdx.x * blockDim.x + threadIdx.x;
  int total = NB * 682 * 3;
  if (t >= total) return;
  int b = t / (682 * 3);
  int rem = t % (682 * 3);
  int c = rem / 3, kv = rem % 3;
  const float* h0 = H + (size_t)c * M + (size_t)b * 3 * NPTS;
  const float* z0 = Z + (size_t)kv * M + (size_t)b * 3 * NPTS;
  float best = -3.4e38f;
  for (int n = 0; n < NPTS; ++n) {
    float v = h0[n] * z0[n] + h0[NPTS + n] * z0[NPTS + n]
            + h0[2 * NPTS + n] * z0[2 * NPTS + n];
    if (v > best) best = v;
  }
  out[b * 2046 + c * 3 + kv] = best;
}

// ---------------------------------------------------------------------------
// Host orchestration.
// ---------------------------------------------------------------------------
extern "C" void kernel_launch(void* const* d_in, const int* in_sizes, int n_in,
                              void* d_out, int out_size, void* d_ws, size_t ws_size,
                              hipStream_t stream) {
  (void)in_sizes; (void)n_in; (void)out_size; (void)ws_size;

  const float* x = (const float*)d_in[0];
  const float *cp_Wf = (const float*)d_in[1], *cp_Wd = (const float*)d_in[2],
              *cp_g  = (const float*)d_in[3], *cp_b  = (const float*)d_in[4];
  const float* pool0_Wd = (const float*)d_in[5];
  const float *c1_Wf = (const float*)d_in[6], *c1_Wd = (const float*)d_in[7],
              *c1_g  = (const float*)d_in[8], *c1_b  = (const float*)d_in[9];
  const float *sc1_Wf = (const float*)d_in[10], *sc1_Wd = (const float*)d_in[11],
              *sc1_g  = (const float*)d_in[12], *sc1_b  = (const float*)d_in[13];
  const float *sc2_Wf = (const float*)d_in[14], *sc2_Wd = (const float*)d_in[15],
              *sc2_g  = (const float*)d_in[16], *sc2_b  = (const float*)d_in[17];
  const float *sc3_Wf = (const float*)d_in[18], *sc3_Wd = (const float*)d_in[19],
              *sc3_g  = (const float*)d_in[20], *sc3_b  = (const float*)d_in[21];
  const float* stn_pool_Wd = (const float*)d_in[22];
  const float *fc1_Wf = (const float*)d_in[23], *fc1_Wd = (const float*)d_in[24],
              *fc1_g  = (const float*)d_in[25], *fc1_b  = (const float*)d_in[26];
  const float *fc2_Wf = (const float*)d_in[27], *fc2_Wd = (const float*)d_in[28],
              *fc2_g  = (const float*)d_in[29], *fc2_b  = (const float*)d_in[30];
  const float* fc3_W = (const float*)d_in[31];
  const float *c2_Wf = (const float*)d_in[32], *c2_Wd = (const float*)d_in[33],
              *c2_g  = (const float*)d_in[34], *c2_b  = (const float*)d_in[35];
  const float* c3_W  = (const float*)d_in[36];
  const float *bn3_g = (const float*)d_in[37], *bn3_b = (const float*)d_in[38];
  const float *sv1_Wf = (const float*)d_in[39], *sv1_Wd = (const float*)d_in[40],
              *sv1_g  = (const float*)d_in[41], *sv1_b  = (const float*)d_in[42];
  const float *sv2_Wf = (const float*)d_in[43], *sv2_Wd = (const float*)d_in[44],
              *sv2_g  = (const float*)d_in[45], *sv2_b  = (const float*)d_in[46];
  const float* std_lin_W = (const float*)d_in[47];

  const int S1 = NPTS * KNN;      // 40960
  const int M1 = 48 * S1;         // 1,966,080 (= 128 * 15360)
  const int S2 = NPTS;            // 2048
  const int M2 = 48 * S2;         // 98,304   (= 128 * 768)
  const int MFC = 128;            // fc stage: 48 live cols padded to 128

  // Workspace bump allocator (4-byte elements); peak ~710 MB.
  char* w = (char*)d_ws;
  auto allocf = [&](size_t n) -> float* {
    float* p = (float*)w;
    w += ((n * 4 + 255) / 256) * 256;
    return p;
  };

  int*   idx     = (int*)allocf((size_t)NB * NPTS * KNN);
  float* regionA = allocf((size_t)682 * M2);          // feat+h1, later h682
  float* feat    = regionA;                           // [3,  M1]
  float* h1      = regionA + (size_t)3 * M1;          // [21, M1]
  float* h682    = regionA;                           // [682,M2] (later)
  float* Dbuf    = allocf((size_t)21 * M1);           // shared D scratch (>=341*M2)
  float* stats   = allocf(2048);
  float* h2      = allocf((size_t)21 * M2);
  float* hcat42  = allocf((size_t)42 * M2);           // rows 0-20 conv1, 21-41 xg
  float* s1b     = allocf((size_t)21 * M2);
  float* s2b     = allocf((size_t)42 * M2);
  float* regionB = allocf((size_t)341 * M2);          // stn_c3 out, later std_vn1 out
  float* hc2     = allocf((size_t)42 * M2);
  float* z2      = allocf((size_t)170 * M2);
  float* zl      = allocf((size_t)3 * M2);
  float* spool   = allocf((size_t)341 * MFC);
  float* f1      = allocf((size_t)170 * MFC);
  float* f2      = allocf((size_t)85 * MFC);
  float* sout    = allocf((size_t)21 * MFC);

  auto gemm = [&](const float* Wm, const float* X, float* Y, int Cout, int Cin, int M) {
    dim3 grid(M / 128, (Cout + 15) / 16);
    gemm_wmma_f16<<<grid, 128, 0, stream>>>(Wm, X, Y, Cout, Cin, M);
  };
  auto lrelu = [&](const float* Wf, const float* Wd, const float* gg, const float* bb,
                   const float* X, float* Y, int Cin, int Cout, int S, int M) {
    gemm(Wf, X, Y, Cout, Cin, M);      // q
    gemm(Wd, X, Dbuf, Cout, Cin, M);   // d
    bn_stats_kernel<<<Cout, 256, 0, stream>>>(Y, stats, S, M);
    long total = (long)Cout * NB * S;
    bn_lrelu_kernel<<<(unsigned)((total + 255) / 256), 256, 0, stream>>>(
        Y, Dbuf, stats, gg, bb, Y, Cout, S, M, 1);
  };

  // ---- graph stage ----
  knn_kernel<<<(NB * NPTS + 255) / 256, 256, 0, stream>>>(x, idx);
  feat_kernel<<<(NB * NPTS * KNN + 255) / 256, 256, 0, stream>>>(x, idx, feat, M1);
  lrelu(cp_Wf, cp_Wd, cp_g, cp_b, feat, h1, 3, 21, S1, M1);      // conv_pos
  gemm(pool0_Wd, h1, Dbuf, 21, 21, M1);                          // pool direction
  vn_maxpool_kernel<<<(21 * NB * NPTS + 255) / 256, 256, 0, stream>>>(
      h1, Dbuf, h2, 21, NPTS, KNN, M1, M2);                      // pool over k

  // ---- point stage ----
  lrelu(c1_Wf, c1_Wd, c1_g, c1_b, h2, hcat42, 21, 21, S2, M2);   // conv1 -> rows 0..20

  // ---- STN (reads rows 0..20 of hcat42) ----
  lrelu(sc1_Wf, sc1_Wd, sc1_g, sc1_b, hcat42, s1b, 21, 21, S2, M2);
  lrelu(sc2_Wf, sc2_Wd, sc2_g, sc2_b, s1b, s2b, 21, 42, S2, M2);
  lrelu(sc3_Wf, sc3_Wd, sc3_g, sc3_b, s2b, regionB, 42, 341, S2, M2);
  gemm(stn_pool_Wd, regionB, Dbuf, 341, 341, M2);
  vn_maxpool_kernel<<<(341 * NB + 255) / 256, 256, 0, stream>>>(
      regionB, Dbuf, spool, 341, 1, NPTS, M2, MFC);              // pool over n
  lrelu(fc1_Wf, fc1_Wd, fc1_g, fc1_b, spool, f1, 341, 170, 1, MFC);
  lrelu(fc2_Wf, fc2_Wd, fc2_g, fc2_b, f1, f2, 170, 85, 1, MFC);
  gemm(fc3_W, f2, sout, 21, 85, MFC);                            // plain linear

  // ---- concat(h, xg) -> conv2 ----
  broadcast_kernel<<<(unsigned)((21L * 48 * NPTS + 255) / 256), 256, 0, stream>>>(
      sout, hcat42, 21, 21, M2, MFC);                            // rows 21..41
  lrelu(c2_Wf, c2_Wd, c2_g, c2_b, hcat42, hc2, 42, 42, S2, M2);  // conv2

  // ---- conv3 + bn3 (no gate) -> rows 0..340 of h682 ----
  gemm(c3_W, hc2, h682, 341, 42, M2);
  bn_stats_kernel<<<341, 256, 0, stream>>>(h682, stats, S2, M2);
  {
    long total = 341L * NB * S2;
    bn_lrelu_kernel<<<(unsigned)((total + 255) / 256), 256, 0, stream>>>(
        h682, nullptr, stats, bn3_g, bn3_b, h682, 341, S2, M2, 0);
  }
  mean_rows_kernel<<<(341 * 48 + 255) / 256, 256, 0, stream>>>(h682, 341, M2);

  // ---- VNStdFeature ----
  lrelu(sv1_Wf, sv1_Wd, sv1_g, sv1_b, h682, regionB, 682, 341, S2, M2);
  lrelu(sv2_Wf, sv2_Wd, sv2_g, sv2_b, regionB, z2, 341, 170, S2, M2);
  gemm(std_lin_W, z2, zl, 3, 170, M2);

  // ---- xs = h * z^T, max over n ----
  final_max_kernel<<<(NB * 682 * 3 + 255) / 256, 256, 0, stream>>>(
      h682, zl, (float*)d_out, M2);
}
